// SelfAttention_27273042330133
// MI455X (gfx1250) — matrix operand order repaired and checked by reading, not compiled, not run on hardware
//
#include <hip/hip_runtime.h>
#include <stdint.h>

typedef __attribute__((ext_vector_type(16))) __bf16 v16bf;
typedef __attribute__((ext_vector_type(8)))  __bf16 v8bf;
typedef __attribute__((ext_vector_type(4)))  __bf16 v4bf;
typedef __attribute__((ext_vector_type(8)))  float  v8f;

#define BLK_M 128
#define BLK_N 128
#define BLK_K 32
#define NTHR  256

// ---- CDNA5 async global->LDS copy (16B per lane), tracked by ASYNCcnt ----
__device__ __forceinline__ void async_copy_b128(const void* gsrc, void* ldst) {
    // VDST = wave-relative LDS byte address == low 32 bits of generic LDS pointer
    asm volatile("global_load_async_to_lds_b128 %0, %1, off"
                 :: "v"((unsigned)(uintptr_t)ldst),
                    "v"((unsigned long long)(uintptr_t)gsrc)
                 : "memory");
}
__device__ __forceinline__ void wait_async() {
    asm volatile("s_wait_asynccnt 0" ::: "memory");
}

__device__ __forceinline__ float fast_rcp(float x) {
    return __builtin_amdgcn_rcpf(x);           // v_rcp_f32, ~1 ulp
}

// ---- one-time weight prep: f32 W[K][N] -> bf16 Wt[N][K] (transposed) ----
__global__ void __launch_bounds__(256)
w_prep(const float* __restrict__ W, __bf16* __restrict__ Wt, int K, int N)
{
    __shared__ __bf16 t[32][33];
    const int n0 = blockIdx.x * 32, k0 = blockIdx.y * 32;
    const int tx = threadIdx.x, ty = threadIdx.y;      // block (32, 8)
    #pragma unroll
    for (int i = ty; i < 32; i += 8)
        t[i][tx] = (__bf16)W[(size_t)(k0 + i) * N + n0 + tx];
    __syncthreads();
    #pragma unroll
    for (int i = ty; i < 32; i += 8)
        Wt[(size_t)(n0 + i) * K + k0 + tx] = t[tx][i];
}

// C[M,N] = act( A[M,K] * Wt[N,K]^T + bias ), bf16 WMMA, f32 accumulate.
// Double-buffered LDS; B (and plain A) staged via async-to-LDS.
// MIX: A comes from f32 x with RWKV time-shift mix a = x*tm + shift(x)*(1-tm)
template<bool MIX>
__global__ void __launch_bounds__(NTHR)
rwkv_gemm(const void* __restrict__ Aptr, const float* __restrict__ tm,
          const __bf16* __restrict__ Wt, const float* __restrict__ bias,
          float* __restrict__ C, int M, int N, int K, int Umask, int act)
{
    __shared__ __bf16 ldsA[2][BLK_M][BLK_K + 8];   // 80B row stride
    __shared__ __bf16 ldsB[2][BLK_N][BLK_K + 8];   // [n][k]

    const int tid  = threadIdx.x;
    const int lane = tid & 31;
    const int wave = tid >> 5;
    const int l15  = lane & 15;
    const int half = lane >> 4;
    const int waveM = (wave & 3) * 32;          // 4 waves tile M
    const int waveN = (wave >> 2) * 64;         // 2 waves tile N
    const int m0 = blockIdx.y * BLK_M;
    const int n0 = blockIdx.x * BLK_N;

    auto stageB = [&](int k0, int buf) {
        #pragma unroll
        for (int i = 0; i < 2; ++i) {
            int c  = tid + i * NTHR;            // 0..511 16B chunks
            int n  = c >> 2;
            int kc = (c & 3) * 8;
            async_copy_b128(Wt + (size_t)(n0 + n) * K + k0 + kc,
                            &ldsB[buf][n][kc]);
        }
    };
    auto stageA = [&](int k0, int buf) {
        if constexpr (MIX) {
            const float* X = (const float*)Aptr;
            #pragma unroll
            for (int i = 0; i < 4; ++i) {
                int c    = tid + i * NTHR;      // 0..1023 float4-chunks
                int row  = c >> 3;
                int kc   = (c & 7) * 4;
                int grow = m0 + row;
                int gk   = k0 + kc;
                const float4 xv = *(const float4*)(X + (size_t)grow * K + gk);
                float4 sh = make_float4(0.f, 0.f, 0.f, 0.f);
                if ((grow & Umask) != 0)        // u==0 -> shifted row is zero
                    sh = *(const float4*)(X + (size_t)(grow - 1) * K + gk);
                const float4 t4 = *(const float4*)(tm + gk);
                v4bf o;
                o[0] = (__bf16)(xv.x * t4.x + sh.x * (1.f - t4.x));
                o[1] = (__bf16)(xv.y * t4.y + sh.y * (1.f - t4.y));
                o[2] = (__bf16)(xv.z * t4.z + sh.z * (1.f - t4.z));
                o[3] = (__bf16)(xv.w * t4.w + sh.w * (1.f - t4.w));
                *(v4bf*)&ldsA[buf][row][kc] = o;
            }
        } else {
            const __bf16* Ab = (const __bf16*)Aptr;
            #pragma unroll
            for (int i = 0; i < 2; ++i) {
                int c   = tid + i * NTHR;       // 0..511 16B chunks
                int row = c >> 2;
                int kc  = (c & 3) * 8;
                async_copy_b128(Ab + (size_t)(m0 + row) * K + k0 + kc,
                                &ldsA[buf][row][kc]);
            }
        }
    };

    v8f acc[2][4];
    const v8f vzero = {};
    #pragma unroll
    for (int t = 0; t < 2; ++t)
        #pragma unroll
        for (int j = 0; j < 4; ++j) acc[t][j] = vzero;

    // ---- prologue: stage tile 0 ----
    stageB(0, 0);
    stageA(0, 0);
    wait_async();
    __syncthreads();

    const int nK = K / BLK_K;
    for (int kt = 0; kt < nK; ++kt) {
        const int cur = kt & 1;
        const int nxt = cur ^ 1;

        // ---- issue next tile's staging (other buffer) before compute ----
        if (kt + 1 < nK) {
            stageB((kt + 1) * BLK_K, nxt);
            stageA((kt + 1) * BLK_K, nxt);
        }

        // ---- fragments + 8 WMMAs on current buffer ----
        v16bf aF[2], bF[4];
        const int kA = half * 8;    // A: elems 0-7 <-> K kA..kA+7, 8-15 <-> kA+16..kA+23
        #pragma unroll
        for (int t = 0; t < 2; ++t) {
            const __bf16* rowp = &ldsA[cur][waveM + t * 16 + l15][0];
            v8bf lo = *(const v8bf*)(rowp + kA);
            v8bf hi = *(const v8bf*)(rowp + kA + 16);
            aF[t] = __builtin_shufflevector(lo, hi, 0,1,2,3,4,5,6,7,8,9,10,11,12,13,14,15);
        }
        const int kB = half * 8;    // B: elems 0-7 <-> K kB..kB+7, 8-15 <-> kB+16..kB+23
        #pragma unroll
        for (int j = 0; j < 4; ++j) {
            const __bf16* colp = &ldsB[cur][waveN + j * 16 + l15][0];
            v8bf lo = *(const v8bf*)(colp + kB);
            v8bf hi = *(const v8bf*)(colp + kB + 16);
            bF[j] = __builtin_shufflevector(lo, hi, 0,1,2,3,4,5,6,7,8,9,10,11,12,13,14,15);
        }
        #pragma unroll
        for (int t = 0; t < 2; ++t)
            #pragma unroll
            for (int j = 0; j < 4; ++j)
                acc[t][j] = __builtin_amdgcn_wmma_f32_16x16x32_bf16(
                    false, aF[t], false, bF[j], (short)0, acc[t][j], false, false);

        // ---- fence next tile (async + LDS stores) once per iteration ----
        if (kt + 1 < nK) {
            wait_async();
            __syncthreads();
        }
    }

    // ---- epilogue: bias (+ optional sigmoid via v_rcp), f32 store ----
    #pragma unroll
    for (int t = 0; t < 2; ++t) {
        #pragma unroll
        for (int j = 0; j < 4; ++j) {
            const int col = n0 + waveN + j * 16 + l15;
            const float bcol = bias[col];
            #pragma unroll
            for (int r = 0; r < 8; ++r) {
                int row = m0 + waveM + t * 16 + half * 8 + r;
                float val = acc[t][j][r] + bcol;
                if (act) val = fast_rcp(1.0f + __expf(-val));
                C[(size_t)row * N + col] = val;
            }
        }
    }
}

// Numerically-stable WKV recurrence; one lane per (b, d); writes sigmoid(r)*wkv as bf16.
__global__ void __launch_bounds__(256)
rwkv_wkv(const float* __restrict__ kk, const float* __restrict__ vv,
         const float* __restrict__ rr, const float* __restrict__ td,
         const float* __restrict__ tf, __bf16* __restrict__ a2,
         int Bn, int Un, int Dn)
{
    int gid = blockIdx.x * blockDim.x + threadIdx.x;
    if (gid >= Bn * Dn) return;
    int b = gid / Dn, d = gid % Dn;
    const float w  = -__expf(td[d]);
    const float u_ = tf[d];
    float aa = 0.f, bb = 0.f, pp = -1e38f;
    size_t base = (size_t)b * Un * Dn + d;
    for (int t = 0; t < Un; ++t) {
        size_t off = base + (size_t)t * Dn;
        float kt = kk[off], vt = vv[off], rt = rr[off];
        float q   = fmaxf(pp, u_ + kt);
        float e1  = __expf(pp - q);
        float e2  = __expf(u_ + kt - q);
        float out = (e1 * aa + e2 * vt) * fast_rcp(e1 * bb + e2);
        float q2  = fmaxf(pp + w, kt);
        float f1  = __expf(pp + w - q2);
        float f2  = __expf(kt - q2);
        aa = f1 * aa + f2 * vt;
        bb = f1 * bb + f2;
        pp = q2;
        a2[off] = (__bf16)(rt * out);
    }
}

extern "C" void kernel_launch(void* const* d_in, const int* in_sizes, int n_in,
                              void* d_out, int out_size, void* d_ws, size_t ws_size,
                              hipStream_t stream) {
    const float* x   = (const float*)d_in[0];
    const float* td  = (const float*)d_in[1];
    const float* tf  = (const float*)d_in[2];
    const float* tmk = (const float*)d_in[3];
    const float* tmv = (const float*)d_in[4];
    const float* tmr = (const float*)d_in[5];
    const float* Wk  = (const float*)d_in[6];
    const float* bk  = (const float*)d_in[7];
    const float* Wv  = (const float*)d_in[8];
    const float* bv  = (const float*)d_in[9];
    const float* Wr  = (const float*)d_in[10];
    const float* br  = (const float*)d_in[11];
    const float* Wo  = (const float*)d_in[12];
    const float* bo  = (const float*)d_in[13];
    float* out = (float*)d_out;

    const int Bc = 8, Uc = 1024, S = 1024, D = 1024;
    const int M = Bc * Uc;                       // 8192
    const size_t MD = (size_t)M * D;

    char* ws = (char*)d_ws;
    float*  kbuf = (float*)(ws);                 // 32 MB
    float*  vbuf = (float*)(ws + MD * 4);        // 32 MB
    float*  rbuf = (float*)(ws + MD * 8);        // 32 MB
    __bf16* a2   = (__bf16*)(ws + MD * 12);      // 16 MB
    __bf16* wtk  = (__bf16*)(ws + MD * 14);      // 2 MB each
    __bf16* wtv  = wtk + (size_t)S * D;
    __bf16* wtr  = wtv + (size_t)S * D;
    __bf16* wto  = wtr + (size_t)S * D;

    // ---- one-time weight convert+transpose ----
    dim3 pblk(32, 8);
    dim3 pgrid(D / 32, S / 32);                  // (32, 32)
    w_prep<<<pgrid, pblk, 0, stream>>>(Wk, wtk, S, D);
    w_prep<<<pgrid, pblk, 0, stream>>>(Wv, wtv, S, D);
    w_prep<<<pgrid, pblk, 0, stream>>>(Wr, wtr, S, D);
    w_prep<<<pgrid, pblk, 0, stream>>>(Wo, wto, D, S);

    dim3 blk(NTHR);
    dim3 grid(D / BLK_N, M / BLK_M);             // (8, 64)

    rwkv_gemm<true><<<grid, blk, 0, stream>>>(x, tmk, wtk, bk, kbuf, M, D, S, Uc - 1, 0);
    rwkv_gemm<true><<<grid, blk, 0, stream>>>(x, tmv, wtv, bv, vbuf, M, D, S, Uc - 1, 0);
    rwkv_gemm<true><<<grid, blk, 0, stream>>>(x, tmr, wtr, br, rbuf, M, D, S, Uc - 1, 1);

    rwkv_wkv<<<(Bc * D + 255) / 256, 256, 0, stream>>>(kbuf, vbuf, rbuf, td, tf, a2,
                                                       Bc, Uc, D);

    dim3 grid2(S / BLK_N, M / BLK_M);            // (8, 64)
    rwkv_gemm<false><<<grid2, blk, 0, stream>>>(a2, nullptr, wto, bo, out, M, S, D, 0x7fffffff, 0);
}
